// EdgeBlock_16449724745525
// MI455X (gfx1250) — matrix operand order repaired and checked by reading, not compiled, not run on hardware
//
#include <hip/hip_runtime.h>

typedef __attribute__((ext_vector_type(16))) __bf16 v16bf;
typedef __attribute__((ext_vector_type(8)))  __bf16 v8bf;
typedef __attribute__((ext_vector_type(8)))  float  v8f;

#define N_EDGES  640000
#define N_NODES  20000
#define N_GRAPHS 16
#define D        64
#define IN_DIM   256
#define LATENT   128
#define OUT_DIM  64

// ws layout (bf16 elements):
//   W1 fragments: 64 frags (k=0..7, n=0..7), each 32 lanes x 16 bf16 = 512 elems
//   W2 fragments: 16 frags (k=0..3, n=0..3), each 512 elems, at element offset 32768
#define W2_ELEM_OFS (64 * 512)

// ---------------------------------------------------------------------------
// Prep: convert f32 weights into WMMA B-fragment-ordered bf16 in workspace.
// B order used here (contiguous K, same as A): lane 0..15 -> column n = Nbase+lane, K = Kbase+0..15
//                           lane 16..31 -> column n = Nbase+(lane-16), K = Kbase+16..31
// packed 2 bf16 per dword in ascending K (t = 0..15).
// ---------------------------------------------------------------------------
__global__ void prep_weights_kernel(const float* __restrict__ W1,
                                    const float* __restrict__ W2,
                                    __bf16* __restrict__ ws) {
    int tid = blockIdx.x * blockDim.x + threadIdx.x;
    if (tid >= 80 * 32) return;
    int frag  = tid >> 5;
    int lane  = tid & 31;
    int khalf = (lane >> 4) & 1;   // 0: K 0..15, 1: K 16..31
    int col16 = lane & 15;
    if (frag < 64) {
        int k = frag >> 3, n = frag & 7;
        __bf16* dst = ws + frag * 512 + lane * 16;
        #pragma unroll
        for (int t = 0; t < 16; ++t) {
            int kk = k * 32 + khalf * 16 + t;
            int nn = n * 16 + col16;
            dst[t] = (__bf16)W1[kk * LATENT + nn];
        }
    } else {
        int f2 = frag - 64;
        int k = f2 >> 2, n = f2 & 3;
        __bf16* dst = ws + W2_ELEM_OFS + f2 * 512 + lane * 16;
        #pragma unroll
        for (int t = 0; t < 16; ++t) {
            int kk = k * 32 + khalf * 16 + t;
            int nn = n * 16 + col16;
            dst[t] = (__bf16)W2[kk * OUT_DIM + nn];
        }
    }
}

// ---------------------------------------------------------------------------
// Main kernel: one 16-edge tile per wave, 8 waves (256 threads) per block.
// ---------------------------------------------------------------------------
__global__ void edge_block_kernel(const float* __restrict__ node_attr,
                                  const float* __restrict__ edge_attr,
                                  const float* __restrict__ global_attr,
                                  const float* __restrict__ node_mask,
                                  const float* __restrict__ edge_mask,
                                  const long long* __restrict__ edge_index,
                                  const long long* __restrict__ eg_index,
                                  const float* __restrict__ b1,
                                  const float* __restrict__ b2,
                                  const __bf16* __restrict__ wbuf,
                                  float* __restrict__ out) {
    extern __shared__ char smem[];
    const int lane = threadIdx.x & 31;
    const int wave = threadIdx.x >> 5;
    const int tile = blockIdx.x * 8 + wave;      // 16 edges per tile

    __bf16* ldsA = (__bf16*)(smem + wave * 8192);   // 16 x 256 bf16 = 8KB per wave

    const long long* senders   = edge_index;
    const long long* receivers = edge_index + N_EDGES;

    // ---- Gather + convert A tile [16 x 256] bf16 into LDS ----
    const int c = lane * 2;                      // 2 f32 per lane per 64-wide segment
    for (int r = 0; r < 16; ++r) {
        const int e = tile * 16 + r;
        const long long s  = senders[e];
        const long long rc = receivers[e];
        const long long g  = eg_index[e];
        const float nms = node_mask[s];
        const float nmr = node_mask[rc];
        float2 ea = *(const float2*)(edge_attr   + (size_t)e  * D + c);
        float2 sa = *(const float2*)(node_attr   + (size_t)s  * D + c);
        float2 ra = *(const float2*)(node_attr   + (size_t)rc * D + c);
        float2 ga = *(const float2*)(global_attr + (size_t)g  * D + c);
        __bf16* row = ldsA + r * IN_DIM;
        row[c]           = (__bf16)ea.x;        row[c + 1]           = (__bf16)ea.y;
        row[D + c]       = (__bf16)(sa.x*nms);  row[D + c + 1]       = (__bf16)(sa.y*nms);
        row[2*D + c]     = (__bf16)(ra.x*nmr);  row[2*D + c + 1]     = (__bf16)(ra.y*nmr);
        row[3*D + c]     = (__bf16)ga.x;        row[3*D + c + 1]     = (__bf16)ga.y;
    }
    asm volatile("s_wait_dscnt 0" ::: "memory");

    // ---- Load A fragments: contiguous K order (same as B), 16x32 per K-step ----
    // lane<16: row=lane, K = Kbase+0..15
    // lane>=16: row=lane-16, K = Kbase+16..31
    const int row16  = lane & 15;
    const int khalf8 = ((lane >> 4) & 1) * 16;   // element offset within K group
    v16bf a[8];
    #pragma unroll
    for (int k = 0; k < 8; ++k) {
        const __bf16* p = ldsA + row16 * IN_DIM + k * 32 + khalf8;
        v8bf lo = *(const v8bf*)p;               // K + 0..7
        v8bf hi = *(const v8bf*)(p + 8);         // K + 8..15
        a[k] = __builtin_shufflevector(lo, hi, 0,1,2,3,4,5,6,7,8,9,10,11,12,13,14,15);
    }

    // ---- GEMM1: [16x256] @ [256x128] -> 8 N-tiles x 8 K-steps = 64 WMMA ----
    v8f acc[8] = {};
    #pragma unroll
    for (int n = 0; n < 8; ++n) {
        #pragma unroll
        for (int k = 0; k < 8; ++k) {
            v16bf b = *(const v16bf*)(wbuf + (k * 8 + n) * 512 + lane * 16);
            acc[n] = __builtin_amdgcn_wmma_f32_16x16x32_bf16(
                false, a[k], false, b, (short)0, acc[n], false, false);
        }
    }

    // ---- bias + relu; write H [16 x 128] bf16 to LDS (reuse A region) ----
    // C layout: lane<16 -> VGPR r = (M=r, N=lane); lane>=16 -> (M=8+r, N=lane-16)
    __bf16* ldsH = ldsA;
    const int mhi = ((lane >> 4) & 1) * 8;
    #pragma unroll
    for (int n = 0; n < 8; ++n) {
        const int col = n * 16 + row16;
        const float bv = b1[col];
        #pragma unroll
        for (int r = 0; r < 8; ++r) {
            const int m = r + mhi;
            float v = acc[n][r] + bv;
            v = v > 0.f ? v : 0.f;
            ldsH[m * LATENT + col] = (__bf16)v;
        }
    }
    asm volatile("s_wait_dscnt 0" ::: "memory");

    // ---- H fragments (A layout, 4 K-steps over K=128) ----
    v16bf h[4];
    #pragma unroll
    for (int k = 0; k < 4; ++k) {
        const __bf16* p = ldsH + row16 * LATENT + k * 32 + khalf8;
        v8bf lo = *(const v8bf*)p;
        v8bf hi = *(const v8bf*)(p + 8);
        h[k] = __builtin_shufflevector(lo, hi, 0,1,2,3,4,5,6,7,8,9,10,11,12,13,14,15);
    }

    // ---- GEMM2: [16x128] @ [128x64] -> 4 N-tiles x 4 K-steps = 16 WMMA ----
    v8f acc2[4] = {};
    #pragma unroll
    for (int n = 0; n < 4; ++n) {
        #pragma unroll
        for (int k = 0; k < 4; ++k) {
            v16bf b = *(const v16bf*)(wbuf + W2_ELEM_OFS + (k * 4 + n) * 512 + lane * 16);
            acc2[n] = __builtin_amdgcn_wmma_f32_16x16x32_bf16(
                false, h[k], false, b, (short)0, acc2[n], false, false);
        }
    }

    // ---- epilogue: bias + edge mask, store f32 ----
    #pragma unroll
    for (int n = 0; n < 4; ++n) {
        const int col = n * 16 + row16;
        const float bv = b2[col];
        #pragma unroll
        for (int r = 0; r < 8; ++r) {
            const int m = r + mhi;
            const int e = tile * 16 + m;
            float v = (acc2[n][r] + bv) * edge_mask[e];
            out[(size_t)e * OUT_DIM + col] = v;
        }
    }
}

// ---------------------------------------------------------------------------
extern "C" void kernel_launch(void* const* d_in, const int* in_sizes, int n_in,
                              void* d_out, int out_size, void* d_ws, size_t ws_size,
                              hipStream_t stream) {
    const float*     node_attr   = (const float*)d_in[0];
    const float*     edge_attr   = (const float*)d_in[1];
    const float*     global_attr = (const float*)d_in[2];
    const float*     node_mask   = (const float*)d_in[3];
    const float*     edge_mask   = (const float*)d_in[4];
    const long long* edge_index  = (const long long*)d_in[5];   // [2, E] int64
    const long long* eg_index    = (const long long*)d_in[6];   // [E]    int64
    const float*     W1          = (const float*)d_in[7];
    const float*     b1          = (const float*)d_in[8];
    const float*     W2          = (const float*)d_in[9];
    const float*     b2          = (const float*)d_in[10];
    float*           out         = (float*)d_out;
    __bf16*          wbuf        = (__bf16*)d_ws;               // 80 KB used

    // 1) weights -> bf16 WMMA B-fragment layout in workspace (80 KB, L2-resident)
    prep_weights_kernel<<<10, 256, 0, stream>>>(W1, W2, wbuf);

    // 2) main fused gather + 2x WMMA-GEMM kernel
    const int tiles  = N_EDGES / 16;     // 40000
    const int blocks = tiles / 8;        // 5000 (exact, no tail)
    edge_block_kernel<<<blocks, 256, 8 * 8192, stream>>>(
        node_attr, edge_attr, global_attr, node_mask, edge_mask,
        edge_index, eg_index, b1, b2, wbuf, out);
}
